// EMACodebook_73924977099398
// MI455X (gfx1250) — compile-verified, matrix operand order checked by reading, not run on hardware
//
#include <hip/hip_runtime.h>

#define DECAY_F 0.99f
#define OMD_F   0.01f
#define EPS_F   1e-5f

#define Bq 16
#define Tq 2048
#define Dq 512
#define Kq 1024
#define Nq (Bq * Tq) /* 32768 */

// LDS layout for the argmin kernel (byte offsets into dynamic LDS, which
// starts at 0 -- no static __shared__ in that kernel):
//   [0      .. 128K) : A_hi   8 waves x 16 rows x 512 bf16
//   [128K   .. 256K) : A_lo
//   [256K   .. 320K) : B tiles, double buffered: 2 x (16 codes x 512 bf16 hi + lo)
#define A_HI_BASE   0u
#define A_LO_BASE   131072u
#define B_BASE      262144u
#define B_BUF_BYTES 32768u        /* 16KB hi + 16KB lo */
#define LDS_TOTAL   327680u       /* 320 KB = CDNA5 per-WG max */

typedef __attribute__((ext_vector_type(16))) __bf16 v16bf;
typedef __attribute__((ext_vector_type(8)))  __bf16 v8bf;
typedef __attribute__((ext_vector_type(8)))  float  v8f;

static __device__ __forceinline__ unsigned int bf_bits(float f) {
    __bf16 b = (__bf16)f;
    return (unsigned int)__builtin_bit_cast(unsigned short, b);
}
static __device__ __forceinline__ float bf_round(float f) {
    __bf16 b = (__bf16)f;
    return (float)b;
}

// ---------------------------------------------------------------------------
// Prep: codebook fp32 -> bf16 hi/lo (L2-resident, 2MB) + ||c||^2 per code.
// ---------------------------------------------------------------------------
__global__ void vq_prep_codebook(const float* __restrict__ cb,
                                 unsigned int* __restrict__ cb_hi,
                                 unsigned int* __restrict__ cb_lo,
                                 float* __restrict__ cnorm) {
    const int k   = blockIdx.x;
    const int tid = threadIdx.x;                  // 256
    const int d0  = tid * 2;
    const float2 c = *(const float2*)(cb + (size_t)k * Dq + d0);
    const float h0 = bf_round(c.x), h1 = bf_round(c.y);
    cb_hi[k * (Dq / 2) + tid] = bf_bits(c.x) | (bf_bits(c.y) << 16);
    cb_lo[k * (Dq / 2) + tid] = bf_bits(c.x - h0) | (bf_bits(c.y - h1) << 16);

    float s = c.x * c.x + c.y * c.y;
    for (int m = 16; m >= 1; m >>= 1) s += __shfl_xor(s, m, 32);
    __shared__ float red[8];
    const int wid = tid >> 5, lane = tid & 31;
    if (lane == 0) red[wid] = s;
    __syncthreads();
    if (tid == 0) {
        float t = 0.f;
        for (int i = 0; i < 8; ++i) t += red[i];
        cnorm[k] = t;
    }
}

// ---------------------------------------------------------------------------
// Zero the scatter accumulators every launch (graph-replay deterministic).
// ---------------------------------------------------------------------------
__global__ void vq_zero(float* __restrict__ sums, float* __restrict__ counts,
                        float* __restrict__ nsum) {
    const int i = blockIdx.x * blockDim.x + threadIdx.x;
    if (i < Kq * Dq) sums[i] = 0.f;
    if (i < Kq)      counts[i] = 0.f;
    if (i == 0)      nsum[0] = 0.f;
}

// ---------------------------------------------------------------------------
// Async-stage one 16-code B tile (hi+lo, 32KB) into LDS buffer `buf`.
// 256 threads x 8 x b128 = 2048 x 16B.  Tracked by ASYNCcnt; caller fences
// with s_wait_asynccnt 0 + barrier before the tile is consumed.
// ---------------------------------------------------------------------------
static __device__ __forceinline__ void stage_btile(const unsigned char* cbhi8,
                                                   const unsigned char* cblo8,
                                                   int kt, int buf, int tid) {
#pragma unroll
    for (int t = 0; t < 4; ++t) {
        const unsigned int off    = (unsigned int)(tid + t * 256) * 16u;  // 0..16K
        const unsigned int ldsHi  = B_BASE + (unsigned int)buf * B_BUF_BYTES + off;
        const unsigned int ldsLo  = ldsHi + 16384u;
        const unsigned long long ghi =
            (unsigned long long)(cbhi8 + (size_t)kt * 16384u + off);
        const unsigned long long glo =
            (unsigned long long)(cblo8 + (size_t)kt * 16384u + off);
        asm volatile("global_load_async_to_lds_b128 %0, %1, off"
                     :: "v"(ldsHi), "v"(ghi) : "memory");
        asm volatile("global_load_async_to_lds_b128 %0, %1, off"
                     :: "v"(ldsLo), "v"(glo) : "memory");
    }
}

// ---------------------------------------------------------------------------
// Distance + argmin via split-bf16 WMMA:  dot(x,c) ~= xh*ch + xh*cl + xl*ch.
// 8 waves/block, 16 rows/wave, 128 rows/block, 256 blocks.
// A staged once per wave into LDS (wave-private slab, no barrier needed).
// B tiles double-buffered in LDS via async global->LDS copies shared by the
// whole workgroup (8x less global codebook traffic); hot loop is pure
// ds_load_b128 + v_wmma.
// ---------------------------------------------------------------------------
__global__ void __launch_bounds__(256)
vq_argmin_wmma(const float* __restrict__ z,
               const unsigned int* __restrict__ cbhi_u,
               const unsigned int* __restrict__ cblo_u,
               const float* __restrict__ cnorm,
               float* __restrict__ out_idx_f,
               int* __restrict__ out_idx_i) {
    extern __shared__ char lds[];
    const int tid  = threadIdx.x;
    const int wave = tid >> 5;
    const int lane = tid & 31;
    const int half = lane >> 4;     // 0: K 0-7/16-23, rows 0-7 ; 1: K 8-15/24-31, rows 8-15
    const int n    = lane & 15;     // A: row M ; B/C/D: column N
    const int rowBase = blockIdx.x * 128 + wave * 16;

    const unsigned char* cbhi8 = (const unsigned char*)cbhi_u;
    const unsigned char* cblo8 = (const unsigned char*)cblo_u;

    // kick off the first B tile while we stage A
    stage_btile(cbhi8, cblo8, /*kt=*/0, /*buf=*/0, tid);

    // ---- stage this wave's 16 rows (contiguous 8192 floats) as bf16 hi/lo ----
    const size_t zbase = (size_t)rowBase * Dq;
    unsigned int* lhi = (unsigned int*)(lds + A_HI_BASE + wave * (16 * Dq * 2));
    unsigned int* llo = (unsigned int*)(lds + A_LO_BASE + wave * (16 * Dq * 2));
    for (int it = 0; it < 128; ++it) {
        const int e2 = it * 32 + lane;                       // pair index
        const float2 x = *(const float2*)(z + zbase + (size_t)e2 * 2);
        const float h0 = bf_round(x.x), h1 = bf_round(x.y);
        lhi[e2] = bf_bits(x.x) | (bf_bits(x.y) << 16);
        llo[e2] = bf_bits(x.x - h0) | (bf_bits(x.y - h1) << 16);
    }

    asm volatile("s_wait_asynccnt 0" ::: "memory");
    __syncthreads();                                 // B tile 0 visible to all

    const __bf16* ahi = (const __bf16*)(lds + A_HI_BASE + wave * (16 * Dq * 2));
    const __bf16* alo = (const __bf16*)(lds + A_LO_BASE + wave * (16 * Dq * 2));

    float bestD[8];
    int   bestI[8];
#pragma unroll
    for (int j = 0; j < 8; ++j) { bestD[j] = 3.4e38f; bestI[j] = 0; }

    union V16 { v16bf v; v8bf h[2]; };
    const int aoff0 = n * Dq + half * 8;   // ISA 16-bit A layout: two 8-elem chunks
    const int boff0 = n * Dq + half * 8;   // local code n within the staged tile

    for (int kt = 0; kt < Kq / 16; ++kt) {
        const int cur = kt & 1;
        if (kt + 1 < Kq / 16)               // prefetch next tile during compute
            stage_btile(cbhi8, cblo8, kt + 1, cur ^ 1, tid);

        const __bf16* bhi = (const __bf16*)(lds + B_BASE + (unsigned)cur * B_BUF_BYTES);
        const __bf16* blo = bhi + 8192;     // lo half of the 32KB buffer

        const int code = kt * 16 + n;
        v8f acc = {};
#pragma unroll 4
        for (int dc = 0; dc < Dq / 32; ++dc) {
            const int db = dc * 32;
            V16 ah, al, bh, bl;
            ah.h[0] = *(const v8bf*)(ahi + aoff0 + db);
            ah.h[1] = *(const v8bf*)(ahi + aoff0 + db + 16);
            al.h[0] = *(const v8bf*)(alo + aoff0 + db);
            al.h[1] = *(const v8bf*)(alo + aoff0 + db + 16);
            bh.h[0] = *(const v8bf*)(bhi + boff0 + db);
            bh.h[1] = *(const v8bf*)(bhi + boff0 + db + 16);
            bl.h[0] = *(const v8bf*)(blo + boff0 + db);
            bl.h[1] = *(const v8bf*)(blo + boff0 + db + 16);
            acc = __builtin_amdgcn_wmma_f32_16x16x32_bf16(false, ah.v, false, bh.v,
                                                          (short)0, acc, false, false);
            acc = __builtin_amdgcn_wmma_f32_16x16x32_bf16(false, ah.v, false, bl.v,
                                                          (short)0, acc, false, false);
            acc = __builtin_amdgcn_wmma_f32_16x16x32_bf16(false, al.v, false, bh.v,
                                                          (short)0, acc, false, false);
        }
        const float cn = cnorm[code];      // same column for all 8 rows in this lane
#pragma unroll
        for (int j = 0; j < 8; ++j) {
            const float dist = cn - 2.0f * acc[j];
            if (dist < bestD[j]) { bestD[j] = dist; bestI[j] = code; }
        }

        asm volatile("s_wait_asynccnt 0" ::: "memory");  // my async stores done
        __syncthreads();               // everyone done reading cur + next staged
    }

    // min-reduce across the 16 lanes that share rows (tie-break: lowest index)
#pragma unroll
    for (int j = 0; j < 8; ++j) {
        float d = bestD[j];
        int   i = bestI[j];
        for (int m = 8; m >= 1; m >>= 1) {
            const float od = __shfl_xor(d, m, 32);
            const int   oi = __shfl_xor(i, m, 32);
            if (od < d || (od == d && oi < i)) { d = od; i = oi; }
        }
        if (n == 0) {
            const int row = rowBase + half * 8 + j;
            out_idx_i[row] = i;
            out_idx_f[row] = (float)i;
        }
    }
}

// ---------------------------------------------------------------------------
// segment_sum via f32 global atomics: one block per token, 4 elems/thread.
// ---------------------------------------------------------------------------
__global__ void vq_scatter(const float* __restrict__ z, const int* __restrict__ idx,
                           float* __restrict__ sums, float* __restrict__ counts) {
    const int t  = blockIdx.x;
    const int id = idx[t];
    const int d0 = threadIdx.x * 4;
    const float4 v = *(const float4*)(z + (size_t)t * Dq + d0);
    float* s = sums + (size_t)id * Dq + d0;
    atomicAdd(s + 0, v.x);
    atomicAdd(s + 1, v.y);
    atomicAdd(s + 2, v.z);
    atomicAdd(s + 3, v.w);
    if (threadIdx.x == 0) atomicAdd(counts + id, 1.0f);
}

// ---------------------------------------------------------------------------
// new_cs = decay*ecs + (1-decay)*counts ; nsum = sum(new_cs). Single block.
// ---------------------------------------------------------------------------
__global__ void vq_newcs(const float* __restrict__ ecs, const float* __restrict__ counts,
                         float* __restrict__ newcs, float* __restrict__ nsum) {
    const int k = threadIdx.x;                 // 1024 threads = 32 waves
    float v = DECAY_F * ecs[k] + OMD_F * counts[k];
    newcs[k] = v;
    for (int m = 16; m >= 1; m >>= 1) v += __shfl_xor(v, m, 32);
    __shared__ float red[32];
    const int wid = k >> 5, lane = k & 31;
    if (lane == 0) red[wid] = v;
    __syncthreads();
    if (wid == 0) {
        float t = red[lane];
        for (int m = 16; m >= 1; m >>= 1) t += __shfl_xor(t, m, 32);
        if (lane == 0) nsum[0] = t;
    }
}

// ---------------------------------------------------------------------------
// codebook_normalized = (decay*ema_w + (1-decay)*sums) / laplace_smoothed
// ---------------------------------------------------------------------------
__global__ void vq_normalize(const float* __restrict__ ew, const float* __restrict__ sums,
                             const float* __restrict__ newcs, const float* __restrict__ nsum,
                             float* __restrict__ cbn) {
    const int k  = blockIdx.x;
    const int d0 = threadIdx.x * 4;
    const float nt = nsum[0];
    const float smoothed = (newcs[k] + EPS_F) / (nt + (float)Kq * EPS_F) * nt;
    const float inv = 1.0f / smoothed;
    const size_t o = (size_t)k * Dq + d0;
    const float4 w = *(const float4*)(ew + o);
    const float4 s = *(const float4*)(sums + o);
    float4 r;
    r.x = (DECAY_F * w.x + OMD_F * s.x) * inv;
    r.y = (DECAY_F * w.y + OMD_F * s.y) * inv;
    r.z = (DECAY_F * w.z + OMD_F * s.z) * inv;
    r.w = (DECAY_F * w.w + OMD_F * s.w) * inv;
    *(float4*)(cbn + o) = r;
}

// ---------------------------------------------------------------------------
// quantized = cbn[idx] ; quantized_st is numerically identical.
// ---------------------------------------------------------------------------
__global__ void vq_gather(const float* __restrict__ cbn, const int* __restrict__ idx,
                          float* __restrict__ qz, float* __restrict__ qst) {
    const int t  = blockIdx.x;
    const int id = idx[t];
    const int d0 = threadIdx.x * 4;
    const float4 q = *(const float4*)(cbn + (size_t)id * Dq + d0);
    *(float4*)(qz  + (size_t)t * Dq + d0) = q;
    *(float4*)(qst + (size_t)t * Dq + d0) = q;
}

extern "C" void kernel_launch(void* const* d_in, const int* in_sizes, int n_in,
                              void* d_out, int out_size, void* d_ws, size_t ws_size,
                              hipStream_t stream) {
    const float* z_e = (const float*)d_in[0];  // [N, D]
    const float* cb  = (const float*)d_in[1];  // [K, D]
    const float* ecs = (const float*)d_in[2];  // [K]
    const float* ew  = (const float*)d_in[3];  // [K, D]
    float* out = (float*)d_out;

    char* ws = (char*)d_ws;
    unsigned int* cb_hi  = (unsigned int*)(ws);                                     // 1 MB
    unsigned int* cb_lo  = (unsigned int*)(ws + (1u << 20));                        // 1 MB
    float*        cnorm  = (float*)(ws + (2u << 20));                               // 4 KB
    float*        counts = (float*)(ws + (2u << 20) + (4u << 10));                  // 4 KB
    float*        newcs  = (float*)(ws + (2u << 20) + (8u << 10));                  // 4 KB
    float*        nsum   = (float*)(ws + (2u << 20) + (12u << 10));                 // 4 B
    int*          idx_i  = (int*)  (ws + (2u << 20) + (16u << 10));                 // 128 KB
    float*        sums   = (float*)(ws + (2u << 20) + (144u << 10));                // 2 MB
    float*        cbn    = (float*)(ws + (4u << 20) + (144u << 10));                // 2 MB

    float* out_idx = out;                              // [N] (indices as f32)
    float* out_q   = out + Nq;                         // [N, D]
    float* out_qst = out + Nq + (size_t)Nq * Dq;       // [N, D]

    vq_prep_codebook<<<Kq, 256, 0, stream>>>(cb, cb_hi, cb_lo, cnorm);
    vq_zero<<<(Kq * Dq + 255) / 256, 256, 0, stream>>>(sums, counts, nsum);
    vq_argmin_wmma<<<Nq / 128, 256, LDS_TOTAL, stream>>>(z_e, cb_hi, cb_lo, cnorm,
                                                         out_idx, idx_i);
    vq_scatter<<<Nq, 128, 0, stream>>>(z_e, idx_i, sums, counts);
    vq_newcs<<<1, 1024, 0, stream>>>(ecs, counts, newcs, nsum);
    vq_normalize<<<Kq, 128, 0, stream>>>(ew, sums, newcs, nsum, cbn);
    vq_gather<<<Nq, 128, 0, stream>>>(cbn, idx_i, out_q, out_qst);
}
